// TCN_20873541058675
// MI455X (gfx1250) — matrix operand order repaired and checked by reading, not compiled, not run on hardware
//
#include <hip/hip_runtime.h>
#include <hip/hip_bf16.h>
#include <math.h>

// ---------------------------------------------------------------------------
// Fused TCN on MI455X (gfx1250, wave32, WMMA, async-LDS-staged implicit GEMM).
//
//   1) pack weights f32->f16 per-tap matrices (A operands, zero-padded cin)
//   2) fold conv-bias + BN into per-channel (alpha, beta)
//   3) convert x -> f16 activation buffer [B][PAD+S][64] (causal zero pad rows)
//   4) 8 fused conv kernels: workgroup stages an 80-row activation tile into
//      LDS via GLOBAL_LOAD_ASYNC_TO_LDS_B128 (ASYNCcnt + s_wait_asynccnt),
//      16 waves run implicit-GEMM with v_wmma_f32_16x16x32_f16; epilogue
//      fuses BN affine (+ds-conv/+residual) + ReLU, packed f16 stores.
//   5) head: sigmoid-attention * linear -> logits (f32) into d_out
//   6) Viterbi (shuffle-parallel over 4 tags) + postprocess -> int32 preds
// ---------------------------------------------------------------------------

typedef _Float16 half_t;
typedef __attribute__((ext_vector_type(2)))  _Float16 v2h;
typedef __attribute__((ext_vector_type(8)))  _Float16 v8h;
typedef __attribute__((ext_vector_type(16))) _Float16 v16h;
typedef __attribute__((ext_vector_type(8)))  float    v8f;

#define S_LEN 2048
#define PAD   16          // >= (K-1)*max_dilation = 16 : causal zero pad rows
#define BATCH 32
#define COUT  256
#define SPAD  (S_LEN + PAD)

// gfx1250 async global->LDS staging (CDNA5 ISA).
// Flip to 0 to fall back to load+ds_store staging.
#define USE_ASYNC_LDS 1

__device__ __forceinline__ v16h cat16(v8h lo, v8h hi) {
  return __builtin_shufflevector(lo, hi, 0,1,2,3,4,5,6,7,8,9,10,11,12,13,14,15);
}

// LDS aperture holds ADDR[63:32]; low 32 bits of a flat LDS pointer are the
// LDS byte offset (ISA §10.2 per-aperture address calculation).
__device__ __forceinline__ unsigned lds_off(const void* p) {
  return (unsigned)(uintptr_t)p;
}

__device__ __forceinline__ void async_copy_b128(half_t* ldst, const half_t* gsrc) {
#if USE_ASYNC_LDS
  unsigned long long ga = (unsigned long long)(uintptr_t)gsrc;
  asm volatile("global_load_async_to_lds_b128 %0, %1, off"
               :: "v"(lds_off(ldst)), "v"(ga) : "memory");
#else
  *(v8h*)ldst = *(const v8h*)gsrc;
#endif
}

__device__ __forceinline__ void async_fence() {
#if USE_ASYNC_LDS
  asm volatile("s_wait_asynccnt 0x0" ::: "memory");
#endif
}

// ---------------------------------------------------------------------------
// Implicit-GEMM conv, one workgroup = 256 cout x 64 timesteps,
// one wave = 16 cout x 64 timesteps (4 16x16 D tiles).
//
// The activation tile (80 rows x CINPAD f16: 64 outputs + 16 history rows for
// causal dilated taps) is staged once into LDS by all 512 threads using
// async-to-LDS b128 ops; every wave then builds its B fragments from
// ds_load_b128 (rows padded by 8 halves so a half-wave's 16 lanes cover all
// 64 banks exactly once -> conflict-free at the b128 minimum).
//
// A fragment (16x32 f16): lane m=lane&15, halves hold K chunks [0..7|16..23]
// or [8..15|24..31] -> two 16B loads from row-major packed weights (global,
// L2-resident, unique per wave, prefetched during the LDS stage).
// B fragment (32x16 f16): lane n(time)=lane&15, same K chunks as A per
// lane -> two 16B LDS loads (+0, +16 halves) from time-major padded rows.
// ---------------------------------------------------------------------------
template<int CINPAD, int TAPS, int DIL, bool HAS_RES, bool HAS_DS>
__global__ __launch_bounds__(512) void conv_wmma(
    const half_t* actIn,               // [B][SPAD][CINPAD] f16, pad rows zero
    const half_t* __restrict__ wpack,  // [TAPS][COUT][CINPAD] f16
    const float*  __restrict__ alpha,  // [COUT] BN scale (folded)
    const float*  __restrict__ beta,   // [COUT] BN shift (conv bias folded)
    const half_t* resIn,               // [B][SPAD][COUT] residual (may alias actOut)
    const half_t* __restrict__ dsIn,   // [B][SPAD][64] (block-0 input) for ds conv
    const half_t* __restrict__ dsW,    // [COUT][64]
    const float*  __restrict__ dsB,    // [COUT] ds bias
    half_t* actOut)                    // [B][SPAD][COUT]
{
  constexpr int LROW   = CINPAD + 8;   // padded LDS row in halves (conflict-free)
  constexpr int ROWS   = 80;           // 16 history rows + 64 output rows
  constexpr int CHUNKS = CINPAD / 8;   // 16B chunks per row
  constexpr int LROWD  = 64 + 8;       // ds tile padded row
  __shared__ half_t sAct[ROWS * LROW];
  __shared__ half_t sDs[HAS_DS ? 64 * LROWD : 1];

  const int lane = threadIdx.x & 31;
  const int wave = threadIdx.x >> 5;
  const int hsel = lane >> 4;      // which half of the wave
  const int l16  = lane & 15;
  const int coutBase = wave * 16;  // 16 waves cover 256 channels
  const int tBase = blockIdx.x * 64;
  const int b = blockIdx.y;

  const half_t* actB = actIn + (size_t)b * SPAD * CINPAD;

  // ---- async stage: global -> LDS (rows tBase-16 .. tBase+63) ----
  {
    const half_t* gBase = actB + (size_t)(PAD + tBase - 16) * CINPAD;
    for (int i = threadIdx.x; i < ROWS * CHUNKS; i += 512) {
      const int row = i / CHUNKS, ch = i % CHUNKS;
      async_copy_b128(&sAct[row * LROW + ch * 8],
                      gBase + (size_t)row * CINPAD + ch * 8);
    }
    if constexpr (HAS_DS) {
      const half_t* gDs = dsIn + ((size_t)b * SPAD + PAD + tBase) * 64;
      for (int i = threadIdx.x; i < 64 * 8; i += 512) {
        const int row = i / 8, ch = i % 8;
        async_copy_b128(&sDs[row * LROWD + ch * 8],
                        gDs + (size_t)row * 64 + ch * 8);
      }
    }
    // hide first A-fragment latency behind the async stage
    __builtin_prefetch(wpack + (size_t)(coutBase + l16) * CINPAD, 0, 3);
    async_fence();           // ASYNCcnt -> 0 before the workgroup barrier
  }
  __syncthreads();

  // ---- WMMA main loop ----
  v8f acc[4] = {};
  for (int k = 0; k < TAPS; ++k) {
    const int tsh = (TAPS - 1 - k) * DIL;               // causal shift of tap k
    const half_t* wk = wpack + (size_t)k * COUT * CINPAD;
    for (int ck = 0; ck < CINPAD / 32; ++ck) {
      const half_t* ap = wk + (size_t)(coutBase + l16) * CINPAD + ck * 32 + 8 * hsel;
      v16h afrag = cat16(*(const v8h*)ap, *(const v8h*)(ap + 16));
#pragma unroll
      for (int j = 0; j < 4; ++j) {                     // reuse A across 4 N-tiles
        const int row = 16 + 16 * j - tsh + l16;        // 0..79 in LDS tile
        const half_t* bp = &sAct[row * LROW + ck * 32 + 8 * hsel];
        v16h bfrag = cat16(*(const v8h*)bp, *(const v8h*)(bp + 16));
        acc[j] = __builtin_amdgcn_wmma_f32_16x16x32_f16(false, afrag, false, bfrag,
                                                        (short)0, acc[j], false, false);
      }
    }
  }

  v8f accD[4] = {};
  if constexpr (HAS_DS) {            // fused 1x1 downsample conv (block 0)
#pragma unroll
    for (int ck = 0; ck < 2; ++ck) {
      const half_t* ap = dsW + (size_t)(coutBase + l16) * 64 + ck * 32 + 8 * hsel;
      v16h afrag = cat16(*(const v8h*)ap, *(const v8h*)(ap + 16));
#pragma unroll
      for (int j = 0; j < 4; ++j) {
        const int row = 16 * j + l16;
        const half_t* bp = &sDs[row * LROWD + ck * 32 + 8 * hsel];
        v16h bfrag = cat16(*(const v8h*)bp, *(const v8h*)(bp + 16));
        accD[j] = __builtin_amdgcn_wmma_f32_16x16x32_f16(false, afrag, false, bfrag,
                                                         (short)0, accD[j], false, false);
      }
    }
  }

  // ---- epilogue: D layout is VGPR r -> M = r + 8*hsel, lane&15 -> N ----
#pragma unroll
  for (int j = 0; j < 4; ++j) {
    const int t = tBase + 16 * j + l16;
    half_t* orow = actOut + ((size_t)b * SPAD + PAD + t) * COUT + coutBase;
    const half_t* rrow = nullptr;
    if constexpr (HAS_RES)
      rrow = resIn + ((size_t)b * SPAD + PAD + t) * COUT + coutBase;
#pragma unroll
    for (int r = 0; r < 8; r += 2) {
      v2h pr;
#pragma unroll
      for (int e = 0; e < 2; ++e) {
        const int cl = r + e + 8 * hsel;
        const int c  = coutBase + cl;
        float v = acc[j][r + e] * alpha[c] + beta[c];
        if constexpr (HAS_DS)  v += accD[j][r + e] + dsB[c];
        if constexpr (HAS_RES) v += (float)rrow[cl];
        pr[e] = (half_t)fmaxf(v, 0.0f);
      }
      *(v2h*)(orow + r + 8 * hsel) = pr;   // paired channels -> 32-bit store
    }
  }
}

// ---------------------------------------------------------------------------
// Small prep kernels
// ---------------------------------------------------------------------------
__global__ void pack_weights(const float* __restrict__ w, half_t* __restrict__ wp,
                             int cout, int cin, int cinPad, int K) {
  int idx = blockIdx.x * blockDim.x + threadIdx.x;
  int total = K * cout * cinPad;
  if (idx >= total) return;
  int c = idx % cinPad;
  int o = (idx / cinPad) % cout;
  int k = idx / (cinPad * cout);
  float v = (c < cin) ? w[((size_t)o * cin + c) * K + k] : 0.0f;
  wp[idx] = (half_t)v;
}

__global__ void prep_affine(const float* __restrict__ g, const float* __restrict__ bb,
                            const float* __restrict__ m, const float* __restrict__ v,
                            const float* __restrict__ convb,
                            float* __restrict__ alpha, float* __restrict__ beta) {
  int c = threadIdx.x;
  if (c >= COUT) return;
  float a = g[c] * rsqrtf(v[c] + 1e-5f);
  alpha[c] = a;
  beta[c]  = (convb[c] - m[c]) * a + bb[c];
}

__global__ void cvt_input(const float* __restrict__ x, half_t* __restrict__ in64) {
  int idx = blockIdx.x * blockDim.x + threadIdx.x;
  int total = BATCH * SPAD * 64;
  if (idx >= total) return;
  int c = idx % 64;
  int t = (idx / 64) % SPAD;
  int b = idx / (64 * SPAD);
  float v = 0.0f;
  if (t >= PAD && c < 34) v = x[((size_t)b * S_LEN + (t - PAD)) * 34 + c];
  in64[idx] = (half_t)v;
}

__global__ void zero_pads(half_t* a, half_t* b) {   // zero pad rows of X and Y
  int idx = blockIdx.x * blockDim.x + threadIdx.x;
  int total = BATCH * PAD * COUT;
  if (idx >= total) return;
  int c = idx % COUT;
  int t = (idx / COUT) % PAD;
  int bb = idx / (COUT * PAD);
  size_t off = ((size_t)bb * SPAD + t) * COUT + c;
  a[off] = (half_t)0.0f;
  b[off] = (half_t)0.0f;
}

// ---------------------------------------------------------------------------
// Head: logits[b,t,j] = sigmoid(z.attn_w+attn_b) * (z.lin_w[j]) + lin_b[j]
// ---------------------------------------------------------------------------
__global__ void head_kernel(const half_t* __restrict__ z,
                            const float* __restrict__ attn_w, const float* __restrict__ attn_b,
                            const float* __restrict__ lin_w,  const float* __restrict__ lin_b,
                            float* __restrict__ logits) {
  int idx = blockIdx.x * blockDim.x + threadIdx.x;
  if (idx >= BATCH * S_LEN) return;
  int t = idx % S_LEN;
  int b = idx / S_LEN;
  const half_t* zr = z + ((size_t)b * SPAD + PAD + t) * COUT;
  float da = 0.f, d0 = 0.f, d1 = 0.f, d2 = 0.f, d3 = 0.f;
  for (int cc = 0; cc < COUT; cc += 8) {
    v8h zc = *(const v8h*)(zr + cc);
#pragma unroll
    for (int e = 0; e < 8; ++e) {
      float zv = (float)zc[e];
      int c = cc + e;
      da += zv * attn_w[c];
      d0 += zv * lin_w[0 * COUT + c];
      d1 += zv * lin_w[1 * COUT + c];
      d2 += zv * lin_w[2 * COUT + c];
      d3 += zv * lin_w[3 * COUT + c];
    }
  }
  float a = 1.0f / (1.0f + expf(-(da + attn_b[0])));
  float* o = logits + (size_t)idx * 4;
  o[0] = a * d0 + lin_b[0];
  o[1] = a * d1 + lin_b[1];
  o[2] = a * d2 + lin_b[2];
  o[3] = a * d3 + lin_b[3];
}

// ---------------------------------------------------------------------------
// Viterbi (NT=4): forward recursion parallelized 4-tags-wide per batch using
// wave32 shuffles (each 4-lane group = one batch, stays inside one wave).
// Backtrack + postprocess are serial per batch on lane j==0 of each group.
// 128 threads = 32 batches x 4 tags.
// ---------------------------------------------------------------------------
__global__ void viterbi_post(const float* __restrict__ logits,
                             const float* __restrict__ start, const float* __restrict__ endp,
                             const float* __restrict__ transIn,
                             unsigned char* __restrict__ hist, int* __restrict__ preds) {
  const int tid = threadIdx.x;
  if (tid >= BATCH * 4) return;
  const int b = tid >> 2;
  const int j = tid & 3;
  const int lane = tid & 31;
  const int gbase = lane & ~3;        // first lane of my 4-lane group

  // trans[i][j] column for my tag j, with CRF overrides
  float trans_j[4];
  for (int i = 0; i < 4; ++i) {
    float v = transIn[i * 4 + j];
    if (i == 1 && j == 2) v = -1000.0f;
    if (i == 2 && j == 0) v = -1000.0f;
    trans_j[i] = v;
  }

  const float* em = logits + (size_t)b * S_LEN * 4;
  float sc = start[j] + em[j];

  for (int t = 1; t < S_LEN; ++t) {
    const float s0 = __shfl(sc, gbase + 0, 32);
    const float s1 = __shfl(sc, gbase + 1, 32);
    const float s2 = __shfl(sc, gbase + 2, 32);
    const float s3 = __shfl(sc, gbase + 3, 32);
    float best = s0 + trans_j[0];
    int arg = 0;
    float v1 = s1 + trans_j[1]; if (v1 > best) { best = v1; arg = 1; }
    float v2 = s2 + trans_j[2]; if (v2 > best) { best = v2; arg = 2; }
    float v3 = s3 + trans_j[3]; if (v3 > best) { best = v3; arg = 3; }
    hist[((size_t)b * S_LEN + t) * 4 + j] = (unsigned char)arg;
    sc = best + em[(size_t)t * 4 + j];
  }

  // final scores -> lane 0 of group picks argmax(sc + end)
  const float f0 = __shfl(sc, gbase + 0, 32) + endp[0];
  const float f1 = __shfl(sc, gbase + 1, 32) + endp[1];
  const float f2 = __shfl(sc, gbase + 2, 32) + endp[2];
  const float f3 = __shfl(sc, gbase + 3, 32) + endp[3];

  __syncthreads();   // make hist stores visible to the backtracking lanes

  if (j != 0) return;

  int last = 0;
  float bestf = f0;
  if (f1 > bestf) { bestf = f1; last = 1; }
  if (f2 > bestf) { bestf = f2; last = 2; }
  if (f3 > bestf) { bestf = f3; last = 3; }

  int* pb = preds + (size_t)b * S_LEN;
  pb[S_LEN - 1] = last;
  int cur = last;
  for (int t = S_LEN - 1; t >= 1; --t) {
    cur = hist[((size_t)b * S_LEN + t) * 4 + cur];
    pb[t - 1] = cur;
  }

  // postprocess: sequential scan with +-2 window (reads updated left neighbors)
  for (int t = 0; t < S_LEN; ++t) {
    if (pb[t] != 3) continue;
    const float* lt = em + (size_t)t * 4;
    float mx = fmaxf(fmaxf(lt[0], lt[1]), fmaxf(lt[2], lt[3]));
    float e0 = expf(lt[0] - mx), e1 = expf(lt[1] - mx);
    float e2 = expf(lt[2] - mx), e3 = expf(lt[3] - mx);
    float p3 = e3 / (e0 + e1 + e2 + e3);
    if (!(p3 < 0.8f)) continue;
    int nonNa = 0, wlen = 0;
    for (int o = -2; o <= 2; ++o) {
      int pos = t + o;
      if (pos >= 0 && pos < S_LEN) { ++wlen; if (pb[pos] != 3) ++nonNa; }
    }
    if (2 * nonNa > wlen) {
      int alt = 0;
      float bv = lt[0];
      if (lt[1] > bv) { bv = lt[1]; alt = 1; }
      if (lt[2] > bv) { bv = lt[2]; alt = 2; }
      pb[t] = alt;
    }
  }
}

// ---------------------------------------------------------------------------
// Host launch
// ---------------------------------------------------------------------------
extern "C" void kernel_launch(void* const* d_in, const int* in_sizes, int n_in,
                              void* d_out, int out_size, void* d_ws, size_t ws_size,
                              hipStream_t stream) {
  (void)in_sizes; (void)n_in; (void)out_size; (void)ws_size;

  // Input index map (setup_inputs insertion order, None leaves dropped):
  // 0:x | block0 @1: w1,b1,w2,b2,bn1(g,b,m,v),bn2(g,b,m,v),ds_w,ds_b (14)
  // blocks1-3 @15/27/39: w1,b1,w2,b2,bn1x4,bn2x4 (12 each)
  // 51:attn_w 52:attn_b 53:lin_w 54:lin_b 55:crf_start 56:crf_end 57:crf_trans
  const float* x = (const float*)d_in[0];
  const int base[4] = {1, 15, 27, 39};
  auto F = [&](int i) { return (const float*)d_in[i]; };

  char* wsb = (char*)d_ws;
  size_t off = 0;
  half_t* in64   = (half_t*)(wsb + off); off += (size_t)BATCH * SPAD * 64  * sizeof(half_t);
  half_t* actX   = (half_t*)(wsb + off); off += (size_t)BATCH * SPAD * COUT * sizeof(half_t);
  half_t* actY   = (half_t*)(wsb + off); off += (size_t)BATCH * SPAD * COUT * sizeof(half_t);
  half_t* wp_c1  = (half_t*)(wsb + off); off += (size_t)3 * COUT * 64  * sizeof(half_t);
  half_t* wp_ds  = (half_t*)(wsb + off); off += (size_t)1 * COUT * 64  * sizeof(half_t);
  half_t* wp_m   = (half_t*)(wsb + off); off += (size_t)7 * 3 * COUT * COUT * sizeof(half_t);
  float*  aff    = (float*)(wsb + off);  off += (size_t)8 * 2 * COUT * sizeof(float);
  unsigned char* hist = (unsigned char*)(wsb + off); off += (size_t)BATCH * S_LEN * 4;

  const size_t WPM = (size_t)3 * COUT * COUT;   // one 3-tap 256x256 weight pack

  // 1) pack weights
  {
    int n = 3 * COUT * 64;
    pack_weights<<<(n + 255) / 256, 256, 0, stream>>>(F(base[0] + 0), wp_c1, COUT, 34, 64, 3);
    n = COUT * 64;
    pack_weights<<<(n + 255) / 256, 256, 0, stream>>>(F(base[0] + 12), wp_ds, COUT, 34, 64, 1);
    // conv2-of-b0 then (conv1,conv2) of b1..b3 -> slots 0..6
    int slot = 0;
    int n2 = 3 * COUT * COUT;
    pack_weights<<<(n2 + 255) / 256, 256, 0, stream>>>(F(base[0] + 2), wp_m + slot++ * WPM, COUT, COUT, COUT, 3);
    for (int blk = 1; blk < 4; ++blk) {
      pack_weights<<<(n2 + 255) / 256, 256, 0, stream>>>(F(base[blk] + 0), wp_m + slot++ * WPM, COUT, COUT, COUT, 3);
      pack_weights<<<(n2 + 255) / 256, 256, 0, stream>>>(F(base[blk] + 2), wp_m + slot++ * WPM, COUT, COUT, COUT, 3);
    }
  }

  // 2) fold conv bias + BN into (alpha, beta); layer L = 2*blk + which
  auto A = [&](int L) { return aff + (size_t)(2 * L + 0) * COUT; };
  auto Bt = [&](int L) { return aff + (size_t)(2 * L + 1) * COUT; };
  for (int blk = 0; blk < 4; ++blk) {
    int bi = base[blk];
    prep_affine<<<1, COUT, 0, stream>>>(F(bi + 4), F(bi + 5), F(bi + 6), F(bi + 7), F(bi + 1),
                                        A(2 * blk + 0), Bt(2 * blk + 0));
    prep_affine<<<1, COUT, 0, stream>>>(F(bi + 8), F(bi + 9), F(bi + 10), F(bi + 11), F(bi + 3),
                                        A(2 * blk + 1), Bt(2 * blk + 1));
  }

  // 3) input -> f16 (with zero pad rows/channels), zero pad rows of X/Y
  {
    int n = BATCH * SPAD * 64;
    cvt_input<<<(n + 255) / 256, 256, 0, stream>>>(x, in64);
    int nz = BATCH * PAD * COUT;
    zero_pads<<<(nz + 255) / 256, 256, 0, stream>>>(actX, actY);
  }

  // 4) fused conv chain (grid: 32 time-tiles x 32 batches, 512 thr = 16 waves)
  dim3 cgrid(S_LEN / 64, BATCH);
  // block 0
  conv_wmma<64, 3, 1, false, false><<<cgrid, 512, 0, stream>>>(
      in64, wp_c1, A(0), Bt(0), nullptr, nullptr, nullptr, nullptr, actX);
  conv_wmma<256, 3, 1, false, true><<<cgrid, 512, 0, stream>>>(
      actX, wp_m + 0 * WPM, A(1), Bt(1), nullptr, in64, wp_ds, F(base[0] + 13), actY);
  // block 1 (d=2)
  conv_wmma<256, 3, 2, false, false><<<cgrid, 512, 0, stream>>>(
      actY, wp_m + 1 * WPM, A(2), Bt(2), nullptr, nullptr, nullptr, nullptr, actX);
  conv_wmma<256, 3, 2, true, false><<<cgrid, 512, 0, stream>>>(
      actX, wp_m + 2 * WPM, A(3), Bt(3), actY, nullptr, nullptr, nullptr, actY);
  // block 2 (d=4)
  conv_wmma<256, 3, 4, false, false><<<cgrid, 512, 0, stream>>>(
      actY, wp_m + 3 * WPM, A(4), Bt(4), nullptr, nullptr, nullptr, nullptr, actX);
  conv_wmma<256, 3, 4, true, false><<<cgrid, 512, 0, stream>>>(
      actX, wp_m + 4 * WPM, A(5), Bt(5), actY, nullptr, nullptr, nullptr, actY);
  // block 3 (d=8)
  conv_wmma<256, 3, 8, false, false><<<cgrid, 512, 0, stream>>>(
      actY, wp_m + 5 * WPM, A(6), Bt(6), nullptr, nullptr, nullptr, nullptr, actX);
  conv_wmma<256, 3, 8, true, false><<<cgrid, 512, 0, stream>>>(
      actX, wp_m + 6 * WPM, A(7), Bt(7), actY, nullptr, nullptr, nullptr, actY);

  // 5) head -> logits (f32) straight into d_out
  float* logits = (float*)d_out;
  {
    int n = BATCH * S_LEN;
    head_kernel<<<(n + 255) / 256, 256, 0, stream>>>(actY, F(51), F(52), F(53), F(54), logits);
  }

  // 6) Viterbi + postprocess -> int32 preds after logits region
  int* preds = (int*)((float*)d_out + (size_t)BATCH * S_LEN * 4);
  viterbi_post<<<1, 128, 0, stream>>>(logits, F(55), F(56), F(57), hist, preds);
}
